// KANLayer_53034256171193
// MI455X (gfx1250) — matrix operand order repaired and checked by reading, not compiled, not run on hardware
//
#include <hip/hip_runtime.h>
#include <math.h>
#include <stdint.h>

// ---------------------------------------------------------------------------
// KAN layer on gfx1250.
//   out = [basis(x) | silu(x)] (2048 x 512*64) @ [coeffs | base_w]  (bf16 WMMA)
// Fast path: pre-pack B-operand to bf16 in d_ws, stream it into LDS with the
// Tensor Data Mover (double buffered, TENSORcnt), 4 features per barrier round.
// ---------------------------------------------------------------------------

typedef __attribute__((ext_vector_type(16))) __bf16 v16bf;
typedef __attribute__((ext_vector_type(8)))  __bf16 v8bf;
typedef __attribute__((ext_vector_type(8)))  float  v8f;
typedef __attribute__((ext_vector_type(4)))  unsigned int u32x4;
typedef __attribute__((ext_vector_type(8)))  int i32x8;
typedef __attribute__((ext_vector_type(4)))  int i32x4;

#define BATCH    2048
#define IN_FEAT  512
#define OUT_FEAT 512
#define NB       60                 // n_basis
#define KPAD     64                 // 60 basis + 1 silu + 3 zero
#define JB       4                  // features staged per barrier round
#define ROUNDS   (IN_FEAT / JB)     // 128
#define BM       64                 // batch tile / workgroup
#define BN       128                // out tile / workgroup
#define NTHREADS 256                // 8 waves, wave32
#define APITCH   72                 // LDS row pitch in bf16 (144B -> 16 distinct banks)
#define TILE_BYTES (JB * BN * KPAD * 2)   // 64 KB per round (packed, unpadded)

#if defined(__HIP_DEVICE_COMPILE__) && \
    __has_builtin(__builtin_amdgcn_tensor_load_to_lds) && \
    __has_builtin(__builtin_amdgcn_s_wait_tensorcnt)
#define HAVE_TDM 1
#else
#define HAVE_TDM 0
#endif

union ABFrag { v16bf v; struct { v8bf lo, hi; } h; };

// A fragment 16x32 bf16: lane<16 -> M=lane, K={0..7,16..23}; lane>=16 -> K={8..15,24..31}
__device__ __forceinline__ v16bf load_a_frag(const __bf16* base, int lane,
                                             int rowBase, int kBase, int pitch) {
  const __bf16* p = base + (rowBase + (lane & 15)) * pitch + kBase + ((lane >> 4) << 3);
  ABFrag f;
  f.h.lo = *(const v8bf*)(p);
  f.h.hi = *(const v8bf*)(p + 16);
  return f.v;
}
// B fragment 32x16 bf16: lane<16 -> N=lane, K={0..7,16..23}; lane>=16 -> K={8..15,24..31} ([o][k] layout)
__device__ __forceinline__ v16bf load_b_frag(const __bf16* base, int lane,
                                             int colBase, int kBase, int pitch) {
  const __bf16* p = base + (colBase + (lane & 15)) * pitch + kBase + ((lane >> 4) << 3);
  ABFrag f;
  f.h.lo = *(const v8bf*)(p);
  f.h.hi = *(const v8bf*)(p + 16);
  return f.v;
}

// Cox-de Boor degree 3 (clamped knots): returns k0, fills N[0..3] = B_{k0..k0+3}(x)
__device__ __forceinline__ int bspline_cubic(float x, const float* Tl,
                                             float t3, float inv_h, float N[4]) {
  int m = (int)floorf((x - t3) * inv_h);
  m = m < 0 ? 0 : (m > 56 ? 56 : m);
  int i = 3 + m;
  if (x < Tl[i] && i > 3) --i;
  if (x >= Tl[i + 1] && i < 59) ++i;
  float left[4], right[4];
  N[0] = 1.0f;
#pragma unroll
  for (int d = 1; d <= 3; ++d) {
    left[d]  = x - Tl[i + 1 - d];
    right[d] = Tl[i + d] - x;
    float saved = 0.0f;
#pragma unroll
    for (int r = 0; r < d; ++r) {
      float tmp = N[r] / (right[r + 1] + left[d - r]);
      N[r] = saved + right[r + 1] * tmp;
      saved = left[d - r] * tmp;
    }
    N[d] = saved;
  }
  return i - 3;
}

__device__ __forceinline__ void stage_a_row(__bf16* row, float x, const float* Tl,
                                            float t3, float inv_h) {
  v8bf z = {};
#pragma unroll
  for (int q = 0; q < KPAD / 8; ++q) *(v8bf*)(row + q * 8) = z;
  float N[4];
  int k0 = bspline_cubic(x, Tl, t3, inv_h, N);   // k0 in [0,56]
  row[k0]     = (__bf16)N[0];
  row[k0 + 1] = (__bf16)N[1];
  row[k0 + 2] = (__bf16)N[2];
  row[k0 + 3] = (__bf16)N[3];
  row[60]     = (__bf16)(x / (1.0f + __expf(-x)));   // fused silu column
}

__device__ __forceinline__ unsigned pack_bf16(float a, float b) {
  union { __bf16 h[2]; unsigned u; } r;
  r.h[0] = (__bf16)a; r.h[1] = (__bf16)b;
  return r.u;
}

// ---------------------------------------------------------------------------
// Pre-pack: ws[oblk][j][o][k] bf16, k: 0..59 coeffs, 60 base_weight, 61..63 zero
// ---------------------------------------------------------------------------
__global__ __launch_bounds__(256) void kan_prepack(
    const float* __restrict__ Coef, const float* __restrict__ Wb,
    unsigned short* __restrict__ ws)
{
  int g = blockIdx.x * 256 + threadIdx.x;   // [0, 4*512*128)
  int oblk = g >> 16;
  int rem  = g & 0xFFFF;
  int j    = rem >> 7;
  int o    = rem & 127;
  int og   = oblk * BN + o;
  const float* src = Coef + ((size_t)j * OUT_FEAT + og) * NB;   // 240B rows, 16B aligned
  unsigned* dst = (unsigned*)(ws + (size_t)g * KPAD);
#pragma unroll
  for (int q = 0; q < 15; ++q) {
    float4 c = *(const float4*)(src + 4 * q);
    dst[2 * q]     = pack_bf16(c.x, c.y);
    dst[2 * q + 1] = pack_bf16(c.z, c.w);
  }
  dst[30] = pack_bf16(Wb[(size_t)j * OUT_FEAT + og], 0.0f);
  dst[31] = 0u;
}

// ---------------------------------------------------------------------------
// TDM issue: 1D tile, data_size=4B, 16384 elems, LDS pad 4 dwords every 32 dwords
// (so each 128B packed row lands on a 144B = APITCH pitch in LDS).
// ---------------------------------------------------------------------------
#if HAVE_TDM
__device__ __forceinline__ void tdm_load_tile(unsigned lds_addr, unsigned long long gaddr) {
  u32x4 g0;
  g0[0] = 1u;                                            // count=1, user descriptor
  g0[1] = lds_addr;
  g0[2] = (unsigned)(gaddr & 0xFFFFFFFFull);
  g0[3] = (unsigned)((gaddr >> 32) & 0x01FFFFFFull) | (2u << 30);  // type=2
  i32x8 g1;
  g1[0] = (int)((2u << 16) | (1u << 20) | (4u << 22) | (3u << 25)); // 4B, pad_en, 32dw interval, 4dw pad
  g1[1] = (int)(((TILE_BYTES / 4) & 0xFFFFu) << 16);     // tensor_dim0[15:0]
  g1[2] = (int)(((TILE_BYTES / 4) >> 16) & 0xFFFF) | (1 << 16);  // tensor_dim0[31:16] | tensor_dim1[15:0]=1
  g1[3] = (int)(((TILE_BYTES / 4) & 0xFFFFu) << 16);     // tensor_dim1[31:16]=0 | tile_dim0
  g1[4] = 1;                                             // tile_dim1=1, tile_dim2=0
  g1[5] = (int)(TILE_BYTES / 4);                         // tensor_dim0_stride[31:0]
  g1[6] = 0;
  g1[7] = 0;
  i32x4 gz4 = {0, 0, 0, 0};
  i32x8 gz8 = {0, 0, 0, 0, 0, 0, 0, 0};
  __builtin_amdgcn_tensor_load_to_lds(g0, g1, gz4, gz4, gz8, 0);
}
#endif

// sync copy of one packed 64KB tile into padded LDS (fallback when no TDM builtin)
__device__ __forceinline__ void stage_b_sync(__bf16* dstLds, const __bf16* src, int tid) {
  for (int i = tid; i < TILE_BYTES / 16; i += NTHREADS) {
    uint4 v = *(const uint4*)((const char*)src + i * 16);
    char* d = (char*)dstLds + i * 16 + (i >> 3) * 16;    // +16B pad per 128B
    *(uint4*)d = v;
  }
}

// ---------------------------------------------------------------------------
// Main kernel (fast path): B from pre-packed ws via TDM, double buffered
// ---------------------------------------------------------------------------
__global__ __launch_bounds__(NTHREADS) void kan_wmma_tdm(
    const float* __restrict__ X,            // (2048, 512)
    const unsigned short* __restrict__ Wpk, // ws: (4, 512, 128, 64) bf16
    const float* __restrict__ Knots,        // (64,)
    float* __restrict__ Out)                // (2048, 512)
{
  __shared__ __bf16 As[JB * BM * APITCH];        // 36 KB
  __shared__ __bf16 Bs[2][JB * BN * APITCH];     // 2 x 72 KB
  __shared__ float  Tl[64];

  const int tid   = threadIdx.x;
  const int lane  = tid & 31;
  const int wid   = tid >> 5;
  const int wRow  = (wid & 1) * 32;
  const int wCol  = (wid >> 1) * 32;
  const int bTile = blockIdx.x * BM;
  const int oblk  = blockIdx.y;
  const int oTile = oblk * BN;

  const size_t slab = (size_t)oblk * IN_FEAT * (BN * KPAD);  // ushort elems
  const unsigned long long gbase =
      (unsigned long long)(uintptr_t)(Wpk + slab);

  if (tid < 64) Tl[tid] = Knots[tid];
#if HAVE_TDM
  if (tid < 32) {  // wave 0 issues round 0's DMA
    tdm_load_tile((unsigned)(uintptr_t)&Bs[0][0], gbase);
  }
#endif
  __syncthreads();
  const float t3    = Tl[3];
  const float inv_h = 1.0f / (Tl[4] - Tl[3]);

  v8f acc00 = {}, acc01 = {}, acc10 = {}, acc11 = {};

  for (int r = 0; r < ROUNDS; ++r) {
    const int j0  = r * JB;
    const int cur = r & 1;

    // ---- stage A: 256 threads, one (jj, b) basis row each
    {
      int jj = tid >> 6, b = tid & 63;
      float x = X[(size_t)(bTile + b) * IN_FEAT + (j0 + jj)];
      stage_a_row(&As[(jj * BM + b) * APITCH], x, Tl, t3, inv_h);
    }

#if HAVE_TDM
    if (tid < 32) {   // wave 0: prefetch next round's B, ensure current arrived
      if (r + 1 < ROUNDS) {
        tdm_load_tile((unsigned)(uintptr_t)&Bs[cur ^ 1][0],
                      gbase + (unsigned long long)(r + 1) * TILE_BYTES);
        __builtin_amdgcn_s_wait_tensorcnt(1);
      } else {
        __builtin_amdgcn_s_wait_tensorcnt(0);
      }
    }
#else
    stage_b_sync(&Bs[cur][0], (const __bf16*)(Wpk + slab) + (size_t)r * (TILE_BYTES / 2), tid);
#endif
    __syncthreads();

    // ---- compute: 4 features x 2 K-steps x 2x2 fragments = 32 WMMA / wave
#pragma unroll
    for (int jj = 0; jj < JB; ++jj) {
      const __bf16* Aj = As + jj * BM * APITCH;
      const __bf16* Bj = &Bs[cur][0] + jj * BN * APITCH;
#pragma unroll
      for (int ks = 0; ks < KPAD; ks += 32) {
        v16bf a0 = load_a_frag(Aj, lane, wRow,      ks, APITCH);
        v16bf a1 = load_a_frag(Aj, lane, wRow + 16, ks, APITCH);
        v16bf b0 = load_b_frag(Bj, lane, wCol,      ks, APITCH);
        v16bf b1 = load_b_frag(Bj, lane, wCol + 16, ks, APITCH);
        acc00 = __builtin_amdgcn_wmma_f32_16x16x32_bf16(false, a0, false, b0,
                                                        (short)0, acc00, false, false);
        acc01 = __builtin_amdgcn_wmma_f32_16x16x32_bf16(false, a0, false, b1,
                                                        (short)0, acc01, false, false);
        acc10 = __builtin_amdgcn_wmma_f32_16x16x32_bf16(false, a1, false, b0,
                                                        (short)0, acc10, false, false);
        acc11 = __builtin_amdgcn_wmma_f32_16x16x32_bf16(false, a1, false, b1,
                                                        (short)0, acc11, false, false);
      }
    }
    __syncthreads();
  }

  // ---- store: C/D layout: VGPR v -> M = v + 8*(lane>=16), N = lane & 15
  const int mOff = (lane >> 4) << 3;
  const int n    = lane & 15;
  float* outBase = Out + (size_t)bTile * OUT_FEAT + oTile;
#pragma unroll
  for (int v = 0; v < 8; ++v) {
    int m = v + mOff;
    outBase[(size_t)(wRow + m)      * OUT_FEAT + (wCol + n)]      = acc00[v];
    outBase[(size_t)(wRow + m)      * OUT_FEAT + (wCol + 16 + n)] = acc01[v];
    outBase[(size_t)(wRow + 16 + m) * OUT_FEAT + (wCol + n)]      = acc10[v];
    outBase[(size_t)(wRow + 16 + m) * OUT_FEAT + (wCol + 16 + n)] = acc11[v];
  }
}

// ---------------------------------------------------------------------------
// Fallback (ws too small): convert fp32 coeffs in-kernel, sync staging
// ---------------------------------------------------------------------------
__global__ __launch_bounds__(NTHREADS) void kan_wmma_fallback(
    const float* __restrict__ X, const float* __restrict__ Coef,
    const float* __restrict__ Wbase, const float* __restrict__ Knots,
    float* __restrict__ Out)
{
  __shared__ __bf16 As[BM * KPAD];
  __shared__ __bf16 Bs[BN * KPAD];
  __shared__ float  Tl[64];

  const int tid   = threadIdx.x;
  const int lane  = tid & 31;
  const int wid   = tid >> 5;
  const int wRow  = (wid & 1) * 32;
  const int wCol  = (wid >> 1) * 32;
  const int bTile = blockIdx.x * BM;
  const int oTile = blockIdx.y * BN;

  if (tid < 64) Tl[tid] = Knots[tid];
  __syncthreads();
  const float t3    = Tl[3];
  const float inv_h = 1.0f / (Tl[4] - Tl[3]);

  v8f acc00 = {}, acc01 = {}, acc10 = {}, acc11 = {};

  for (int j = 0; j < IN_FEAT; ++j) {
    const float* Cj = Coef + ((size_t)j * OUT_FEAT + oTile) * NB;
    for (int idx = tid; idx < BN * (NB / 4); idx += NTHREADS) {
      int o = idx / (NB / 4);
      int q = idx - o * (NB / 4);
      float4 c4 = *(const float4*)(Cj + o * NB + 4 * q);
      unsigned* dst = (unsigned*)&Bs[o * KPAD + 4 * q];
      dst[0] = pack_bf16(c4.x, c4.y);
      dst[1] = pack_bf16(c4.z, c4.w);
    }
    if (tid < BN) {
      unsigned* dst = (unsigned*)&Bs[tid * KPAD + 60];
      dst[0] = pack_bf16(Wbase[(size_t)j * OUT_FEAT + oTile + tid], 0.0f);
      dst[1] = 0u;
    }
    if (tid < BM) {
      float x = X[(size_t)(bTile + tid) * IN_FEAT + j];
      stage_a_row(&As[tid * KPAD], x, Tl, t3, inv_h);
    }
    if (j + 1 < IN_FEAT) {
      const float* Pn = Coef + ((size_t)(j + 1) * OUT_FEAT + oTile) * NB;
      __builtin_prefetch(Pn + tid * 32, 0, 1);
    }
    __syncthreads();
#pragma unroll
    for (int ks = 0; ks < KPAD; ks += 32) {
      v16bf a0 = load_a_frag(As, lane, wRow,      ks, KPAD);
      v16bf a1 = load_a_frag(As, lane, wRow + 16, ks, KPAD);
      v16bf b0 = load_b_frag(Bs, lane, wCol,      ks, KPAD);
      v16bf b1 = load_b_frag(Bs, lane, wCol + 16, ks, KPAD);
      acc00 = __builtin_amdgcn_wmma_f32_16x16x32_bf16(false, a0, false, b0,
                                                      (short)0, acc00, false, false);
      acc01 = __builtin_amdgcn_wmma_f32_16x16x32_bf16(false, a0, false, b1,
                                                      (short)0, acc01, false, false);
      acc10 = __builtin_amdgcn_wmma_f32_16x16x32_bf16(false, a1, false, b0,
                                                      (short)0, acc10, false, false);
      acc11 = __builtin_amdgcn_wmma_f32_16x16x32_bf16(false, a1, false, b1,
                                                      (short)0, acc11, false, false);
    }
    __syncthreads();
  }

  const int mOff = (lane >> 4) << 3;
  const int n    = lane & 15;
  float* outBase = Out + (size_t)bTile * OUT_FEAT + oTile;
#pragma unroll
  for (int v = 0; v < 8; ++v) {
    int m = v + mOff;
    outBase[(size_t)(wRow + m)      * OUT_FEAT + (wCol + n)]      = acc00[v];
    outBase[(size_t)(wRow + m)      * OUT_FEAT + (wCol + 16 + n)] = acc01[v];
    outBase[(size_t)(wRow + 16 + m) * OUT_FEAT + (wCol + n)]      = acc10[v];
    outBase[(size_t)(wRow + 16 + m) * OUT_FEAT + (wCol + 16 + n)] = acc11[v];
  }
}

extern "C" void kernel_launch(void* const* d_in, const int* in_sizes, int n_in,
                              void* d_out, int out_size, void* d_ws, size_t ws_size,
                              hipStream_t stream) {
  (void)in_sizes; (void)n_in; (void)out_size;
  const float* x      = (const float*)d_in[0];
  const float* coeffs = (const float*)d_in[1];
  const float* wbase  = (const float*)d_in[2];
  const float* knots  = (const float*)d_in[3];
  float* out = (float*)d_out;

  const size_t need = (size_t)(OUT_FEAT / BN) * IN_FEAT * BN * KPAD * 2;  // 32 MB
  dim3 grid(BATCH / BM, OUT_FEAT / BN);   // 32 x 4

  if (ws_size >= need && d_ws != nullptr) {
    kan_prepack<<<(OUT_FEAT / BN) * IN_FEAT * BN / 256, 256, 0, stream>>>(
        coeffs, wbase, (unsigned short*)d_ws);
    kan_wmma_tdm<<<grid, NTHREADS, 0, stream>>>(
        x, (const unsigned short*)d_ws, knots, out);
  } else {
    kan_wmma_fallback<<<grid, NTHREADS, 0, stream>>>(x, coeffs, wbase, knots, out);
  }
}
